// OTKernel_1726576854100
// MI455X (gfx1250) — matrix operand order repaired and checked by reading, not compiled, not run on hardware
//
#include <hip/hip_runtime.h>
#include <hip/hip_bf16.h>

// ---------------------------------------------------------------------------
// Problem constants (fixed by reference): n=64, S=1024, D=128, M=4, O=64
//   B = n*M = 256 sinkhorn tiles of [1024 x 64]
// Workspace layout (bytes):
//   [0,            16MB)  xb   : x in bf16            (64*1024*128)
//   [16MB,         +64KB) wbuf : weight in bf16       (4*64*128)
//   [16MB+64KB,    +64MB) K    : f32 K tiles, 256 x 256KB
//                               upper 128KB of each tile reused for T^T (bf16)
// ---------------------------------------------------------------------------

typedef __attribute__((ext_vector_type(16))) __bf16        v16bf;
typedef __attribute__((ext_vector_type(8)))  float         v8f;
typedef __attribute__((ext_vector_type(8)))  unsigned int  v8u;
typedef __attribute__((ext_vector_type(4)))  unsigned int  v4u;

__device__ __forceinline__ unsigned short f2bf(float f) {
    unsigned int u = __float_as_uint(f);
    u += 0x7FFFu + ((u >> 16) & 1u);           // round-to-nearest-even
    return (unsigned short)(u >> 16);
}
__device__ __forceinline__ float bf2f(unsigned short h) {
    return __uint_as_float(((unsigned int)h) << 16);
}

__device__ __forceinline__ v8f wmma_bf16(v8u a, v8u b, v8f c) {
    return __builtin_amdgcn_wmma_f32_16x16x32_bf16(
        /*neg_a=*/false, __builtin_bit_cast(v16bf, a),
        /*neg_b=*/false, __builtin_bit_cast(v16bf, b),
        /*c_mod=*/(short)0, c, /*reuse_a=*/false, /*reuse_b=*/false);
}

// A-fragment pair-dword index for 16-bit A 16x32: VGPR r, lane-half h
//   K = (r<4 ? 2r+8h : 16+2(r-4)+8h)  ->  dword pair index K/2
__device__ __forceinline__ int a_pidx(int r, int h) {
    return r + 4 * h + (r >= 4 ? 4 : 0);
}

// --- CDNA5 async global->LDS DMA (ASYNCcnt-tracked, in-order completion) ----
// Generic shared pointers keep the wave LDS byte offset in addr[31:0].
__device__ __forceinline__ unsigned lds_off(const void* p) {
    return (unsigned)(unsigned long long)p;
}
__device__ __forceinline__ void async_b128(unsigned lds_byte, const void* g) {
    asm volatile("global_load_async_to_lds_b128 %0, %1, off"
                 :: "v"(lds_byte), "v"(g) : "memory");
}
__device__ __forceinline__ void wait_async_6() {
    asm volatile("s_wait_asynccnt 0x6" ::: "memory");   // older group done
}
__device__ __forceinline__ void wait_async_0() {
    asm volatile("s_wait_asynccnt 0x0" ::: "memory");
}

// --- CDNA5 Tensor Data Mover: 2-D bf16 tile, global -> padded LDS ----------
// D# built per CDNA5 ISA (§8). Rows of rowlen*2 bytes are
// deposited with a 4-dword pad every 64 dwords (pad_interval=5, pad_amount=3)
// => LDS row stride 68 dwords, matching the conflict-free WMMA tile layout.
// Must be issued by ONE wave (branch-guarded: TDM ignores EXEC).
__device__ __forceinline__ void tdm_load_tile_2d_bf16(
    unsigned lds_byte, const void* gaddr,
    unsigned rowlen, unsigned nrows, unsigned tensor_rows, unsigned row_stride) {
    unsigned long long ga = (unsigned long long)gaddr;
    v4u g0;
    g0.x = 1u;                                       // count=1, user descriptor
    g0.y = lds_byte;                                 // lds_addr [63:32]
    g0.z = (unsigned)(ga & 0xFFFFFFFFu);             // global_addr lo
    g0.w = (unsigned)((ga >> 32) & 0x01FFFFFFu)      // global_addr hi [120:96]
         | (2u << 30);                               // type=2 ("image")
    v8u g1;
    g1[0] = (1u << 16)                               // data_size: 2 bytes
          | (1u << 20)                               // pad_enable
          | (5u << 22)                               // pad_interval: 64 dwords
          | (3u << 25);                              // pad_amount: 4 dwords
    g1[1] = (rowlen & 0xFFFFu) << 16;                // tensor_dim0[15:0] @63:48
    g1[2] = (rowlen >> 16) | ((tensor_rows & 0xFFFFu) << 16); // dim0 hi | dim1 lo
    g1[3] = (tensor_rows >> 16) | (rowlen << 16);    // dim1 hi | tile_dim0
    g1[4] = nrows;                                   // tile_dim1 | tile_dim2=0
    g1[5] = row_stride;                              // tensor_dim0_stride lo
    g1[6] = 0u;                                      // stride0 hi | stride1 lo
    g1[7] = 0u;
    v4u gz = {0u, 0u, 0u, 0u};                       // groups 2/3: unused dims
    asm volatile("tensor_load_to_lds %0, %1, %2, %3"
                 :: "s"(g0), "s"(g1), "s"(gz), "s"(gz) : "memory");
}

// ---------------------------------------------------------------------------
// Phase 0: f32 -> bf16 conversion of x and weight
// ---------------------------------------------------------------------------
__global__ __launch_bounds__(256) void cvt_bf16_kernel(
    const float* __restrict__ x, const float* __restrict__ w,
    unsigned short* __restrict__ xb, unsigned short* __restrict__ wb) {
    const int NX = 64 * 1024 * 128;            // 8388608
    const int NW = 4 * 64 * 128;               // 32768
    int i = blockIdx.x * 256 + threadIdx.x;
    if (i < NX)            xb[i] = f2bf(x[i]);
    else if (i < NX + NW)  wb[i - NX] = f2bf(w[i - NX]);
}

// ---------------------------------------------------------------------------
// Phase 1: K[b][s][o] = sum_d x[n][s][d] * w[m][o][d]   (b = n*4+m)
// One workgroup (256 thr / 8 waves) per (b, 128-row s-block): grid = 2048.
// A tile DMA'd by the Tensor Data Mover (one descriptor, hardware padding);
// W^T tile transposed through ds_store.
// ---------------------------------------------------------------------------
__global__ __launch_bounds__(256) void gemm_k_kernel(
    const unsigned short* __restrict__ xb,
    const unsigned short* __restrict__ wb,
    float* __restrict__ Kbuf) {
    __shared__ unsigned int   ldsA[128 * 68];    // 34816 B, stride 68 dw/row
    __shared__ unsigned short ldsWT[128 * 68];   // 17408 B (d-major, 68 halfs/row)

    const int gid = blockIdx.x;
    const int b = gid >> 3, sb = gid & 7;
    const int n = b >> 2,  m  = b & 3;
    const int tid = threadIdx.x;

    // --- A tile: xb[n][sb*128 .. +128][0..127] -> ldsA via one TDM descriptor
    if (tid < 32) {
        const unsigned short* gA = xb + (size_t)(n * 1024 + sb * 128) * 128;
        tdm_load_tile_2d_bf16(lds_off(ldsA), gA,
                              /*rowlen=*/128, /*nrows=*/128,
                              /*tensor_rows=*/1024, /*row_stride=*/128);
    }
    // --- W tile transposed: wb[m][o][d] -> ldsWT[d][o]
    const unsigned int* wg = (const unsigned int*)wb + m * 4096; // [o][d/2] dwords
#pragma unroll
    for (int i = 0; i < 16; ++i) {
        int idx = tid + i * 256;
        int o = idx >> 6, dp = idx & 63;
        unsigned int v = wg[idx];
        ldsWT[(2 * dp)     * 68 + o] = (unsigned short)(v & 0xFFFFu);
        ldsWT[(2 * dp + 1) * 68 + o] = (unsigned short)(v >> 16);
    }
    if (tid < 32) __builtin_amdgcn_s_wait_tensorcnt(0);  // TDM drained (wave 0)
    __syncthreads();

    const int wid = tid >> 5, lane = tid & 31;
    const int st = wid;                          // 16-row subtile within s-block
    const int mrow = lane & 15, h = lane >> 4;
    const unsigned int* wtd = (const unsigned int*)ldsWT; // stride 34 dw per d-row

    v8f acc[4] = {};
#pragma unroll
    for (int kt = 0; kt < 4; ++kt) {             // K = 128 in steps of 32
        v8u a;
        int abase = (st * 16 + mrow) * 68 + kt * 16;
#pragma unroll
        for (int r = 0; r < 8; ++r) a[r] = ldsA[abase + a_pidx(r, h)];
        int d = kt * 32;                         // B: K row base; column = lane % 16
#pragma unroll
        for (int ot = 0; ot < 4; ++ot) {
            v8u bf;
#pragma unroll
            for (int r = 0; r < 8; ++r) bf[r] = (unsigned int)ldsWT[(d + 2 * a_pidx(r, h)) * 68 + ot * 16 + mrow] | ((unsigned int)ldsWT[(d + 2 * a_pidx(r, h) + 1) * 68 + ot * 16 + mrow] << 16);
            acc[ot] = wmma_bf16(a, bf, acc[ot]);
        }
    }
    // --- store D: lane/VGPR j -> (s = 16*st + j + 8h, o = 16*ot + lane%16)
    float* Kout = Kbuf + (size_t)b * 65536;
#pragma unroll
    for (int ot = 0; ot < 4; ++ot) {
        int o = ot * 16 + (lane & 15);
#pragma unroll
        for (int j = 0; j < 8; ++j) {
            int s = sb * 128 + st * 16 + j + 8 * h;
            Kout[s * 64 + o] = acc[ot][j];
        }
    }
}

// ---------------------------------------------------------------------------
// Phase 2: log-domain Sinkhorn, 100 iters, entirely in LDS.
// One workgroup (1024 thr) per tile b. K tile kept as bf16 in LDS (132KB).
// Writes T^T (bf16, [o][s]) into the upper half of tile b's K slot.
// ---------------------------------------------------------------------------
__global__ __launch_bounds__(1024) void sinkhorn_kernel(
    const float* __restrict__ Kbuf, char* __restrict__ kregion) {
    __shared__ unsigned short Kt[1024 * 66];     // 132KB, stride 66 halfs/row
    __shared__ float us[1024];
    __shared__ float pm[1024];
    __shared__ float vs[64];
    __shared__ float cm[64];

    const int b = blockIdx.x, t = threadIdx.x;
    const float* Kb = Kbuf + (size_t)b * 65536;
#pragma unroll 4
    for (int i = 0; i < 64; ++i) {               // 65536 f32 -> bf16 in LDS
        int idx = t + i * 1024;
        Kt[(idx >> 6) * 66 + (idx & 63)] = f2bf(Kb[idx]);
    }
    us[t] = 0.0f;
    if (t < 64) vs[t] = 0.0f;
    __syncthreads();

    const float a_log = -2.7725887222397812f;    // log(64/1024)
    const float eps = 1.0f, inv_eps = 1.0f;
    const int c = t & 63, ch = t >> 6, r0 = ch << 6;

    for (int it = 0; it < 100; ++it) {
        // ---- row pass: lse over o (64), update u[t]
        float u = us[t];
        float mx = -3.0e38f;
        for (int o = 0; o < 64; ++o)
            mx = fmaxf(mx, (bf2f(Kt[t * 66 + o]) + u + vs[o]) * inv_eps);
        float sm = 0.0f;
        for (int o = 0; o < 64; ++o)
            sm += __expf((bf2f(Kt[t * 66 + o]) + u + vs[o]) * inv_eps - mx);
        us[t] = eps * (a_log - (mx + __logf(sm))) + u;
        __syncthreads();

        // ---- col pass: lse over s (1024), chunked 16x64, update v[c]
        float v = vs[c];
        float lm = -3.0e38f;
        for (int r = 0; r < 64; ++r)
            lm = fmaxf(lm, (bf2f(Kt[(r0 + r) * 66 + c]) + us[r0 + r] + v) * inv_eps);
        pm[t] = lm;
        __syncthreads();
        if (t < 64) {
            float m2 = -3.0e38f;
#pragma unroll
            for (int j = 0; j < 16; ++j) m2 = fmaxf(m2, pm[j * 64 + t]);
            cm[t] = m2;
        }
        __syncthreads();
        float M = cm[c], ls = 0.0f;
        for (int r = 0; r < 64; ++r)
            ls += __expf((bf2f(Kt[(r0 + r) * 66 + c]) + us[r0 + r] + v) * inv_eps - M);
        pm[t] = ls;
        __syncthreads();
        if (t < 64) {
            float s2 = 0.0f;
#pragma unroll
            for (int j = 0; j < 16; ++j) s2 += pm[j * 64 + t];
            vs[t] = -eps * (cm[t] + __logf(s2)) + vs[t];
        }
        __syncthreads();
    }

    // ---- T^T[o][s] = exp((K+u+v)/eps), bf16, into upper half of this tile
    unsigned short* Tt = (unsigned short*)(kregion + (size_t)b * 262144 + 131072);
    float u = us[t];
    for (int o = 0; o < 64; ++o) {
        float Tv = __expf((bf2f(Kt[t * 66 + o]) + u + vs[o]) * inv_eps);
        Tt[o * 1024 + t] = f2bf(Tv);
    }
}

// ---------------------------------------------------------------------------
// Phase 3: out[n][o][m*128+d] = sum_s T^T[b][o][s] * x[n][s][d]
// One workgroup (256 thr / 8 waves) per b. M=64(o), N=128(d), K=1024(s).
// Double-buffered LDS slabs fed by async-LDS DMA, overlapped with WMMA.
// Each thread issues exactly 6 async B128 ops per slab; loads complete in
// order, so  s_wait_asynccnt 6  drains only the previous slab's group.
// ---------------------------------------------------------------------------
__global__ __launch_bounds__(256) void gemm_out_kernel(
    const unsigned short* __restrict__ xb,
    const char* __restrict__ kregion,
    float* __restrict__ out) {
    __shared__ unsigned int Ta[2][64 * 36];      // 2 x  9216 B (stride 36 dw)
    __shared__ unsigned int Xb[2][64 * 68];      // 2 x 17408 B (stride 68 dw)

    const int b = blockIdx.x;
    const int n = b >> 2, m = b & 3;
    const int tid = threadIdx.x, wid = tid >> 5, lane = tid & 31;
    const int mt = wid & 3, nt0 = (wid >> 2) * 4;
    const int mrow = lane & 15, h = lane >> 4;

    const uint4* Tg = (const uint4*)(kregion + (size_t)b * 262144 + 131072);
    const uint4* Xg = (const uint4*)xb + (size_t)n * 1024 * 16;

    auto issue_slab = [&](int kb, int bufi) {
        uint4* Ta4 = (uint4*)Ta[bufi];           // row stride 9 uint4
        uint4* Xb4 = (uint4*)Xb[bufi];           // row stride 17 uint4
#pragma unroll
        for (int i = 0; i < 2; ++i) {            // A slab: 64 rows x 8 uint4
            int idx4 = tid + i * 256;
            int row = idx4 >> 3, c4 = idx4 & 7;
            async_b128(lds_off(Ta4 + row * 9 + c4), Tg + row * 128 + kb * 8 + c4);
        }
#pragma unroll
        for (int i = 0; i < 4; ++i) {            // B slab: 64 rows x 16 uint4
            int idx4 = tid + i * 256;
            int row = idx4 >> 4, c4 = idx4 & 15;
            async_b128(lds_off(Xb4 + row * 17 + c4), Xg + (kb * 64 + row) * 16 + c4);
        }
    };

    v8f acc[4] = {};
    issue_slab(0, 0);
    for (int kb = 0; kb < 16; ++kb) {
        const int cur = kb & 1;
        if (kb + 1 < 16) {
            issue_slab(kb + 1, 1 - cur);         // prefetch next slab (async)
            wait_async_6();                      // wait only for slab kb
        } else {
            wait_async_0();
        }
        __syncthreads();                         // slab kb visible to all waves
        if (kb + 2 < 16)                         // gfx1250 global_prefetch_b8
            __builtin_prefetch(&Tg[(tid >> 1) * 128 + (kb + 2) * 8], 0, 1);

        const unsigned int* TaC = Ta[cur];
        const unsigned int* XbC = Xb[cur];
#pragma unroll
        for (int ks = 0; ks < 2; ++ks) {         // two K=32 steps per slab
            v8u a;
            int abase = (mt * 16 + mrow) * 36 + ks * 16;
#pragma unroll
            for (int r = 0; r < 8; ++r) a[r] = TaC[abase + a_pidx(r, h)];
            int krow = ks * 32;
#pragma unroll
            for (int i = 0; i < 4; ++i) {
                v8u bf;
#pragma unroll
                for (int r = 0; r < 8; ++r)
                    bf[r] = (unsigned int)((const unsigned short*)XbC)[(krow + 2 * a_pidx(r, h)) * 136 + (nt0 + i) * 16 + mrow] | ((unsigned int)((const unsigned short*)XbC)[(krow + 2 * a_pidx(r, h) + 1) * 136 + (nt0 + i) * 16 + mrow] << 16);
                acc[i] = wmma_bf16(a, bf, acc[i]);
            }
        }
        __syncthreads();                         // all reads of slab kb done
    }
#pragma unroll
    for (int i = 0; i < 4; ++i) {
        int d = (nt0 + i) * 16 + (lane & 15);
#pragma unroll
        for (int j = 0; j < 8; ++j) {
            int o = mt * 16 + j + 8 * h;
            out[(size_t)n * 32768 + o * 512 + m * 128 + d] = acc[i][j];
        }
    }
}

// ---------------------------------------------------------------------------
extern "C" void kernel_launch(void* const* d_in, const int* in_sizes, int n_in,
                              void* d_out, int out_size, void* d_ws, size_t ws_size,
                              hipStream_t stream) {
    (void)in_sizes; (void)n_in; (void)out_size; (void)ws_size;
    const float* x = (const float*)d_in[0];     // [64,1024,128]
    const float* w = (const float*)d_in[1];     // [4,64,128]
    float* out = (float*)d_out;                 // [64,64,512]

    char* ws = (char*)d_ws;
    unsigned short* xb = (unsigned short*)ws;                    // 16 MB
    unsigned short* wb = (unsigned short*)(ws + 16777216);       // 64 KB
    char* kreg = ws + 16842752;                                  // 64 MB
    float* Kbuf = (float*)kreg;

    cvt_bf16_kernel<<<(8421376 + 255) / 256, 256, 0, stream>>>(x, w, xb, wb);
    gemm_k_kernel<<<2048, 256, 0, stream>>>(xb, wb, Kbuf);
    sinkhorn_kernel<<<256, 1024, 0, stream>>>(Kbuf, kreg);
    gemm_out_kernel<<<256, 256, 0, stream>>>(xb, kreg, out);
}
